// PatchAttentionNet_45612552683812
// MI455X (gfx1250) — compile-verified, matrix operand order checked by reading, not run on hardware
//
#include <hip/hip_runtime.h>
#include <hip/hip_bf16.h>
#include <math.h>

typedef __attribute__((ext_vector_type(16))) _Float16 v16h;
typedef __attribute__((ext_vector_type(8)))  float    v8f;

#define Bb   4
#define Nn_  256
#define Oo   5
#define KK2  25      // K*K patch positions
#define Ff   256
#define Hh   4
#define Dd   64
#define DFFc 1024

constexpr int TOK  = Bb * Nn_ * Oo * KK2;   // 128000 tokens
constexpr int ROWS = Bb * Nn_ * Oo;         // 5120 output rows
constexpr int PB   = Nn_ * Oo * Ff;         // 327680 elems per batch for LN

constexpr int KPANEL  = 256;                // B panel depth staged in LDS
constexpr int BSTRIDE = KPANEL + 8;         // +16B pad: breaks LDS bank conflicts

// ---------------------------------------------------------------------------
// helpers
// ---------------------------------------------------------------------------
__device__ __forceinline__ float wave_max32(float v) {
  #pragma unroll
  for (int o = 16; o > 0; o >>= 1) v = fmaxf(v, __shfl_xor(v, o, 32));
  return v;
}
__device__ __forceinline__ float wave_sum32(float v) {
  #pragma unroll
  for (int o = 16; o > 0; o >>= 1) v += __shfl_xor(v, o, 32);
  return v;
}

union FragU { v16h v; uint4 q[2]; };

// Load one f16 16x32 fragment given per-lane base (row/col and half-offset
// already folded in): elements 0..7 at +k, elements 8..15 at +k+16.
__device__ __forceinline__ v16h frag_at(const _Float16* p) {
  FragU f;
  f.q[0] = *(const uint4*)(p);
  f.q[1] = *(const uint4*)(p + 16);
  return f.v;
}

// ---------------------------------------------------------------------------
// GEMM: C[M][N] = A[M][K] * W[N][K]^T (+bias) (+addsrc) (relu?) -> f32/f16
// block = 256 threads = 8 waves; wave computes 32 rows x 64 cols (2 A frags
// share 4 B frags -> 8 WMMAs per k-step); block tile is 256x64. The B panel
// (64 cols x KPANEL) is staged in LDS with async-to-LDS copies shared by all
// 8 waves; A fragments are register double-buffered.
// grid = (M/256, N/64). M%256==0, N%64==0, K%KPANEL==0 guaranteed by caller.
//
// f16 16x32 fragment layout (A and B identical): lane L<16 -> row/col L,
// elements 0..7 = K [k0+off, k0+off+8), elements 8..15 = +16; off = 8*(L>=16).
// Both runs are contiguous -> two b128 loads per fragment.
// ---------------------------------------------------------------------------
__global__ __launch_bounds__(256) void gemm_f16_wmma(
    const _Float16* __restrict__ A, const _Float16* __restrict__ W,
    const float* __restrict__ bias, const float* __restrict__ addsrc,
    float* __restrict__ outF, _Float16* __restrict__ outH,
    int M, int N, int K, int doRelu)
{
  __shared__ _Float16 Bs[64 * BSTRIDE];     // ~33KB

  const int lane    = threadIdx.x & 31;
  const int wave    = threadIdx.x >> 5;
  const int rowBase = (blockIdx.x * 8 + wave) * 32;
  const int colBase = blockIdx.y * 64;

  const int n0    = lane & 15;              // row (A) / col (B) within tile
  const int half8 = ((lane >> 4) & 1) << 3; // K sub-offset per lane half

  const _Float16* aPtr0 = A + (size_t)(rowBase + n0) * K + half8;
  const _Float16* aPtr1 = aPtr0 + (size_t)16 * K;

  v8f acc[8];
  #pragma unroll
  for (int j = 0; j < 8; ++j) acc[j] = (v8f){};

  // preload first A fragments (register double buffer, 2 row groups)
  FragU aN0, aN1;
  aN0.q[0] = *(const uint4*)(aPtr0);
  aN0.q[1] = *(const uint4*)(aPtr0 + 16);
  aN1.q[0] = *(const uint4*)(aPtr1);
  aN1.q[1] = *(const uint4*)(aPtr1 + 16);

  // async staging assignment: 4 threads per B row, 8 b128 chunks each
  const int srow = threadIdx.x >> 2;          // 0..63
  const int skc  = (threadIdx.x & 3) << 3;    // 0,8,16,24
  const _Float16* srcRow = W + (size_t)(colBase + srow) * K + skc;
  const unsigned ldsBase = (unsigned)(uintptr_t)&Bs[srow * BSTRIDE + skc];

  for (int kp = 0; kp < K; kp += KPANEL) {
    __syncthreads();   // previous panel fully consumed before overwrite

    // --- async stage of B panel: 64 rows x KPANEL f16 -> LDS (b128 chunks)
    {
      const _Float16* src = srcRow + kp;
      unsigned lds = ldsBase;
      #pragma unroll
      for (int c = 0; c < KPANEL / 32; ++c) { // 8 chunks, stride 32 elems
        asm volatile("global_load_async_to_lds_b128 %0, %1, off"
                     :: "v"(lds), "v"(src) : "memory");
        src += 32;
        lds += 64;
      }
    }
    asm volatile("s_wait_asynccnt 0x0" ::: "memory");
    __syncthreads();

    // --- K panel: 8 steps of 32; A double-buffered, B frags from LDS
    #pragma unroll
    for (int kk = 0; kk < KPANEL; kk += 32) {
      const v16h a0 = aN0.v;
      const v16h a1 = aN1.v;
      const int kNext = kp + kk + 32;
      if (kNext < K) {                      // prefetch next A fragments
        aN0.q[0] = *(const uint4*)(aPtr0 + kNext);
        aN0.q[1] = *(const uint4*)(aPtr0 + kNext + 16);
        aN1.q[0] = *(const uint4*)(aPtr1 + kNext);
        aN1.q[1] = *(const uint4*)(aPtr1 + kNext + 16);
      }
      // load all 4 B fragments first (separate regs -> one ds clause)
      const _Float16* bb = &Bs[n0 * BSTRIDE + kk + half8];
      const v16h b0 = frag_at(bb);
      const v16h b1 = frag_at(bb + 16 * BSTRIDE);
      const v16h b2 = frag_at(bb + 32 * BSTRIDE);
      const v16h b3 = frag_at(bb + 48 * BSTRIDE);
      acc[0] = __builtin_amdgcn_wmma_f32_16x16x32_f16(false, a0, false, b0, (short)0, acc[0], false, false);
      acc[1] = __builtin_amdgcn_wmma_f32_16x16x32_f16(false, a0, false, b1, (short)0, acc[1], false, false);
      acc[2] = __builtin_amdgcn_wmma_f32_16x16x32_f16(false, a0, false, b2, (short)0, acc[2], false, false);
      acc[3] = __builtin_amdgcn_wmma_f32_16x16x32_f16(false, a0, false, b3, (short)0, acc[3], false, false);
      acc[4] = __builtin_amdgcn_wmma_f32_16x16x32_f16(false, a1, false, b0, (short)0, acc[4], false, false);
      acc[5] = __builtin_amdgcn_wmma_f32_16x16x32_f16(false, a1, false, b1, (short)0, acc[5], false, false);
      acc[6] = __builtin_amdgcn_wmma_f32_16x16x32_f16(false, a1, false, b2, (short)0, acc[6], false, false);
      acc[7] = __builtin_amdgcn_wmma_f32_16x16x32_f16(false, a1, false, b3, (short)0, acc[7], false, false);
    }
  }

  // C/D layout: VGPR r -> (M=r, N=lane) for lanes 0-15, (M=r+8, N=lane-16)
  const int r0 = (lane >> 4) * 8;
  #pragma unroll
  for (int g = 0; g < 2; ++g) {             // two 16-row groups
    #pragma unroll
    for (int j = 0; j < 4; ++j) {
      const int col = colBase + 16 * j + n0;
      const float bv = bias ? bias[col] : 0.0f;
      const v8f av = acc[g * 4 + j];
      #pragma unroll
      for (int r = 0; r < 8; ++r) {
        const int row = rowBase + g * 16 + r0 + r;
        float v = av[r] + bv;
        if (doRelu) v = fmaxf(v, 0.0f);
        const size_t idx = (size_t)row * N + col;
        if (addsrc) v += addsrc[idx];
        if (outF) outF[idx] = v;
        if (outH) outH[idx] = (_Float16)v;
      }
    }
  }
}

// ---------------------------------------------------------------------------
// Attention: one wave per (b,n,o,h) unit. qkv rows are tokens of 768 f16
// ([q 256 | k 256 | v 256]); 25 tokens per (b,n,o) patch, center at p=12.
// ---------------------------------------------------------------------------
__global__ __launch_bounds__(256) void attention_kernel(
    const _Float16* __restrict__ qkv, _Float16* __restrict__ merged)
{
  __shared__ float wsm[8][32];
  const int lane = threadIdx.x & 31;
  const int wave = threadIdx.x >> 5;
  const int unit = blockIdx.x * 8 + wave;   // 0 .. ROWS*H-1
  const int h    = unit & (Hh - 1);
  const int bno  = unit >> 2;
  const size_t t0 = (size_t)bno * KK2;

  const _Float16* qc = qkv + (t0 + 12) * 768 + h * Dd;         // center query
  const _Float16* kb = qkv + t0 * 768 + 256 + h * Dd;
  const _Float16* vb = qkv + t0 * 768 + 512 + h * Dd;

  // lanes 0..24: one score each (dot over 64 dims), scaled by 1/sqrt(64)
  float score = -1e30f;
  if (lane < KK2) {
    const _Float16* kp = kb + (size_t)lane * 768;
    float s = 0.f;
    #pragma unroll 8
    for (int d = 0; d < Dd; ++d) s += (float)qc[d] * (float)kp[d];
    score = s * 0.125f;
  }
  const float m  = wave_max32(score);
  const float e  = (lane < KK2) ? __expf(score - m) : 0.f;
  const float sm = wave_sum32(e);
  wsm[wave][lane] = e / sm;
  __syncthreads();

  // each lane accumulates 2 of the 64 output dims
  const int d0 = lane * 2;
  float a0 = 0.f, a1 = 0.f;
  #pragma unroll
  for (int p = 0; p < KK2; ++p) {
    const float wp = wsm[wave][p];
    a0 += wp * (float)vb[(size_t)p * 768 + d0];
    a1 += wp * (float)vb[(size_t)p * 768 + d0 + 1];
  }
  _Float16* out = merged + (size_t)bno * (Hh * Dd) + h * Dd + d0;
  out[0] = (_Float16)a0;
  out[1] = (_Float16)a1;
}

// ---------------------------------------------------------------------------
// LayerNorm over (N,O,F) per batch element: reduce then apply
// ---------------------------------------------------------------------------
__global__ __launch_bounds__(256) void ln_reduce_kernel(
    const float* __restrict__ x, float* __restrict__ stats)
{
  __shared__ float sA[256], sB[256];
  const int b = blockIdx.y;
  const float* xb = x + (size_t)b * PB;
  float s = 0.f, s2 = 0.f;
  for (int i = blockIdx.x * blockDim.x + threadIdx.x; i < PB;
       i += gridDim.x * blockDim.x) {
    const float v = xb[i];
    s += v; s2 += v * v;
  }
  sA[threadIdx.x] = s; sB[threadIdx.x] = s2;
  __syncthreads();
  for (int st = 128; st > 0; st >>= 1) {
    if (threadIdx.x < st) {
      sA[threadIdx.x] += sA[threadIdx.x + st];
      sB[threadIdx.x] += sB[threadIdx.x + st];
    }
    __syncthreads();
  }
  if (threadIdx.x == 0) {
    atomicAdd(&stats[2 * b + 0], sA[0]);
    atomicAdd(&stats[2 * b + 1], sB[0]);
  }
}

__global__ __launch_bounds__(256) void ln_apply_kernel(
    const float* __restrict__ x, const float* __restrict__ g,
    const float* __restrict__ beta, const float* __restrict__ stats,
    float* __restrict__ outF, _Float16* __restrict__ outH, int total)
{
  for (int i = blockIdx.x * blockDim.x + threadIdx.x; i < total;
       i += gridDim.x * blockDim.x) {
    const int b = i / PB;
    const int r = i - b * PB;
    const float mu  = stats[2 * b + 0] * (1.0f / PB);
    const float var = stats[2 * b + 1] * (1.0f / PB) - mu * mu;
    const float v = (x[i] - mu) * rsqrtf(var + 1e-5f) * g[r] + beta[r];
    if (outF) outF[i] = v;
    if (outH) outH[i] = (_Float16)v;
  }
}

// ---------------------------------------------------------------------------
// small utility kernels
// ---------------------------------------------------------------------------
__global__ void cvt_f32_to_f16(const float* __restrict__ s,
                               _Float16* __restrict__ d, int n) {
  for (int i = blockIdx.x * blockDim.x + threadIdx.x; i < n;
       i += gridDim.x * blockDim.x)
    d[i] = (_Float16)s[i];
}
__global__ void copy_f32(const float* __restrict__ s, float* __restrict__ d, int n) {
  for (int i = blockIdx.x * blockDim.x + threadIdx.x; i < n;
       i += gridDim.x * blockDim.x)
    d[i] = s[i];
}
__global__ void zero_f32(float* __restrict__ p, int n) {
  if ((int)threadIdx.x < n) p[threadIdx.x] = 0.f;
}

// ---------------------------------------------------------------------------
extern "C" void kernel_launch(void* const* d_in, const int* in_sizes, int n_in,
                              void* d_out, int out_size, void* d_ws, size_t ws_size,
                              hipStream_t stream) {
  const float* x    = (const float*)d_in[0];
  const float* Wq   = (const float*)d_in[1];
  const float* bq   = (const float*)d_in[2];
  const float* Wk   = (const float*)d_in[3];
  const float* bk   = (const float*)d_in[4];
  const float* Wv   = (const float*)d_in[5];
  const float* bv   = (const float*)d_in[6];
  const float* Wf   = (const float*)d_in[7];
  const float* bf   = (const float*)d_in[8];
  const float* ln1w = (const float*)d_in[9];
  const float* ln1b = (const float*)d_in[10];
  const float* ln2w = (const float*)d_in[11];
  const float* ln2b = (const float*)d_in[12];
  const float* W1   = (const float*)d_in[13];
  const float* b1   = (const float*)d_in[14];
  const float* W2   = (const float*)d_in[15];
  const float* b2   = (const float*)d_in[16];

  char* ws = (char*)d_ws;
  size_t off = 0;
  auto alloc = [&](size_t bytes) -> void* {
    void* p = ws + off;
    off += (bytes + 255) & ~(size_t)255;
    return p;
  };

  _Float16* xb      = (_Float16*)alloc((size_t)TOK * Ff * 2);      // x in f16
  _Float16* wqkv    = (_Float16*)alloc((size_t)768 * Ff * 2);      // [Wq;Wk;Wv]
  _Float16* wfH     = (_Float16*)alloc((size_t)Ff * Ff * 2);
  _Float16* w1H     = (_Float16*)alloc((size_t)DFFc * Ff * 2);
  _Float16* w2H     = (_Float16*)alloc((size_t)Ff * DFFc * 2);
  float*    bqkv    = (float*)   alloc(768 * 4);                   // [bq;bk;bv]
  _Float16* qkvH    = (_Float16*)alloc((size_t)TOK * 768 * 2);     // fused qkv
  _Float16* mergedH = (_Float16*)alloc((size_t)ROWS * Ff * 2);
  float*    attent  = (float*)   alloc((size_t)ROWS * Ff * 4);
  float*    out1F   = (float*)   alloc((size_t)ROWS * Ff * 4);
  _Float16* out1H   = (_Float16*)alloc((size_t)ROWS * Ff * 2);
  _Float16* hidH    = (_Float16*)alloc((size_t)ROWS * DFFc * 2);
  float*    ytmp    = (float*)   alloc((size_t)ROWS * Ff * 4);
  float*    stats   = (float*)   alloc(256);                       // 2 sets of 4x2

  // --- prep: zero LN stats, convert inputs/weights to f16, pack fused bias
  zero_f32<<<1, 32, 0, stream>>>(stats, 16);
  cvt_f32_to_f16<<<2048, 256, 0, stream>>>(x,  xb,   TOK * Ff);
  cvt_f32_to_f16<<<64,   256, 0, stream>>>(Wq, wqkv,                   Ff * Ff);
  cvt_f32_to_f16<<<64,   256, 0, stream>>>(Wk, wqkv + (size_t)Ff*Ff,   Ff * Ff);
  cvt_f32_to_f16<<<64,   256, 0, stream>>>(Wv, wqkv + (size_t)2*Ff*Ff, Ff * Ff);
  cvt_f32_to_f16<<<64,   256, 0, stream>>>(Wf, wfH, Ff * Ff);
  cvt_f32_to_f16<<<256,  256, 0, stream>>>(W1, w1H, DFFc * Ff);
  cvt_f32_to_f16<<<256,  256, 0, stream>>>(W2, w2H, Ff * DFFc);
  copy_f32<<<1, 256, 0, stream>>>(bq, bqkv,        Ff);
  copy_f32<<<1, 256, 0, stream>>>(bk, bqkv + Ff,   Ff);
  copy_f32<<<1, 256, 0, stream>>>(bv, bqkv + 2*Ff, Ff);

  // --- fused QKV projection: (128000 x 256) @ (256 x 768) -> f16
  {
    dim3 g(TOK / 256, 768 / 64);
    gemm_f16_wmma<<<g, 256, 0, stream>>>(xb, wqkv, bqkv, nullptr,
                                         nullptr, qkvH, TOK, 768, Ff, 0);
  }

  // --- local patch attention (softmax over 25, per head) -> merged f16
  attention_kernel<<<(ROWS * Hh) / 8, 256, 0, stream>>>(qkvH, mergedH);

  // --- output projection: (5120 x 256) @ (256 x 256) -> attent f32
  {
    dim3 g(ROWS / 256, Ff / 64);
    gemm_f16_wmma<<<g, 256, 0, stream>>>(mergedH, wfH, bf, nullptr,
                                         attent, nullptr, ROWS, Ff, Ff, 0);
  }

  // --- LayerNorm 1 (per batch over N,O,F) -> out1 (f32 + f16)
  ln_reduce_kernel<<<dim3(40, Bb), 256, 0, stream>>>(attent, stats);
  ln_apply_kernel<<<2560, 256, 0, stream>>>(attent, ln1w, ln1b, stats,
                                            out1F, out1H, Bb * PB);

  // --- FFN: relu(out1 @ W1^T + b1) @ W2^T + b2 + out1
  {
    dim3 g(ROWS / 256, DFFc / 64);
    gemm_f16_wmma<<<g, 256, 0, stream>>>(out1H, w1H, b1, nullptr,
                                         nullptr, hidH, ROWS, DFFc, Ff, 1);
  }
  {
    dim3 g(ROWS / 256, Ff / 64);
    gemm_f16_wmma<<<g, 256, 0, stream>>>(hidH, w2H, b2, out1F,
                                         ytmp, nullptr, ROWS, Ff, DFFc, 0);
  }

  // --- LayerNorm 2 -> d_out (f32)
  ln_reduce_kernel<<<dim3(40, Bb), 256, 0, stream>>>(ytmp, stats + 8);
  ln_apply_kernel<<<2560, 256, 0, stream>>>(ytmp, ln2w, ln2b, stats + 8,
                                            (float*)d_out, nullptr, Bb * PB);
}
